// AttnDecoder_15350213116181
// MI455X (gfx1250) — matrix operand order repaired and checked by reading, not compiled, not run on hardware
//
#include <hip/hip_runtime.h>
#include <hip/hip_bf16.h>

#define Hh   512
#define Ss   128
#define Bb   32
#define Tt   128
#define Vv   32000
#define BOS  1

typedef unsigned short ushort_t;
typedef __attribute__((ext_vector_type(16))) __bf16 v16bf;
typedef __attribute__((ext_vector_type(8)))  float  v8f;

union BfTile { v16bf v; uint4 q[2]; };

__device__ __forceinline__ ushort_t f2bf(float f) {
    unsigned u = __float_as_uint(f);
    u += 0x7fffu + ((u >> 16) & 1u);          // round-to-nearest-even
    return (ushort_t)(u >> 16);
}

// ---------------- one-time preprocessing ----------------

__global__ void cvt_f32_bf16(const float* __restrict__ src, ushort_t* __restrict__ dst, int n) {
    int i = blockIdx.x * blockDim.x + threadIdx.x;
    int stride = gridDim.x * blockDim.x;
    for (; i < n; i += stride) dst[i] = f2bf(src[i]);
}

__global__ void init_hidden(const float* __restrict__ enc_hid,
                            float* __restrict__ h, ushort_t* __restrict__ hbf) {
    int i = blockIdx.x * blockDim.x + threadIdx.x;   // 0..16383
    float v = enc_hid[i];
    h[i] = v;
    hbf[i] = f2bf(v);
}

__global__ void copy_hidden_out(const float* __restrict__ h, float* __restrict__ out_tail) {
    int i = blockIdx.x * blockDim.x + threadIdx.x;
    out_tail[i] = h[i];
}

// ---------------- step kernel 1: attention (1 block per batch row) ----------------

__global__ __launch_bounds__(256) void attn_step(
    const float* __restrict__ enc,       // [B,S,H]
    const float* __restrict__ h,         // [B,H]
    const int*   __restrict__ targets,   // [B,T]
    const float* __restrict__ emb_tab,   // [V,H]
    const float* __restrict__ W_attn,    // [H,H]
    ushort_t*    __restrict__ xgru_bf,   // [B,2H] bf16  = [emb, ctx]
    float*       __restrict__ ctx_f,     // [B,H]
    int t)
{
    __shared__ float hs[Hh];
    __shared__ float qs[Hh];
    __shared__ float es[Ss];
    __shared__ float tr[Ss];

    const int b   = blockIdx.x;
    const int tid = threadIdx.x;

    hs[tid]       = h[b * Hh + tid];
    hs[tid + 256] = h[b * Hh + 256 + tid];
    __syncthreads();

    // q = h @ W_attn^T : each thread 2 rows of W_attn (K contiguous)
    #pragma unroll
    for (int jj = 0; jj < 2; ++jj) {
        const int j = tid + jj * 256;
        const float4* wr = (const float4*)(W_attn + (size_t)j * Hh);
        const float4* hv = (const float4*)hs;
        float s = 0.f;
        for (int k4 = 0; k4 < Hh / 4; ++k4) {
            float4 w = wr[k4], x = hv[k4];
            s += w.x * x.x + w.y * x.y + w.z * x.z + w.w * x.w;
        }
        qs[j] = s;
    }
    __syncthreads();

    // energy[s] = q . enc[b,s,:]
    if (tid < Ss) {
        const float4* er = (const float4*)(enc + ((size_t)b * Ss + tid) * Hh);
        const float4* qv = (const float4*)qs;
        float s = 0.f;
        for (int k4 = 0; k4 < Hh / 4; ++k4) {
            float4 e = er[k4], q = qv[k4];
            s += e.x * q.x + e.y * q.y + e.z * q.z + e.w * q.w;
        }
        es[tid] = s;
    }
    __syncthreads();

    // softmax over S=128
    if (tid < Ss) tr[tid] = es[tid];
    __syncthreads();
    for (int off = Ss / 2; off > 0; off >>= 1) {
        if (tid < off) tr[tid] = fmaxf(tr[tid], tr[tid + off]);
        __syncthreads();
    }
    const float mx = tr[0];
    __syncthreads();
    if (tid < Ss) { float e = expf(es[tid] - mx); es[tid] = e; tr[tid] = e; }
    __syncthreads();
    for (int off = Ss / 2; off > 0; off >>= 1) {
        if (tid < off) tr[tid] += tr[tid + off];
        __syncthreads();
    }
    const float inv = 1.0f / tr[0];
    __syncthreads();
    if (tid < Ss) es[tid] *= inv;
    __syncthreads();

    // context[j] = sum_s attn[s] * enc[b,s,j]   (coalesced over j)
    float c0 = 0.f, c1 = 0.f;
    for (int s = 0; s < Ss; ++s) {
        const float a = es[s];
        const float* er = enc + ((size_t)b * Ss + s) * Hh;
        c0 += a * er[tid];
        c1 += a * er[tid + 256];
    }

    // embedding of teacher-forced input token
    const int tok = (t == 0) ? BOS : targets[b * Tt + (t - 1)];
    const float e0 = emb_tab[(size_t)tok * Hh + tid];
    const float e1 = emb_tab[(size_t)tok * Hh + 256 + tid];

    ushort_t* xr = xgru_bf + (size_t)b * (2 * Hh);
    xr[tid]            = f2bf(e0);
    xr[256 + tid]      = f2bf(e1);
    xr[Hh + tid]       = f2bf(c0);
    xr[Hh + 256 + tid] = f2bf(c1);
    ctx_f[b * Hh + tid]       = c0;
    ctx_f[b * Hh + 256 + tid] = c1;
}

// ---------------- step kernel 2: gate GEMMs via WMMA ----------------
// Gx = Xg[32,1024] @ Wih^T + b_ih ;  Gh = Hb[32,512] @ Whh^T + b_hh   (both [32,1536])

__global__ __launch_bounds__(256) void gates_gemm(
    const ushort_t* __restrict__ Xg,   // [32,1024] bf16
    const ushort_t* __restrict__ Hb,   // [32,512]  bf16
    const ushort_t* __restrict__ Wih,  // [1536,1024] bf16 (row-major, K contiguous)
    const ushort_t* __restrict__ Whh,  // [1536,512]  bf16
    const float* __restrict__ b_ih, const float* __restrict__ b_hh,
    float* __restrict__ Gx, float* __restrict__ Gh)
{
    const int wave = threadIdx.x >> 5;
    const int lane = threadIdx.x & 31;
    const int n0   = (blockIdx.x * 8 + wave) * 16;    // < 1536
    const int nlo  = lane & 15;
    const int kh   = lane >> 4;
    const int aoff = kh * 8;

    v8f ax0 = {}, ax1 = {}, ah0 = {}, ah1 = {};

    {   // X @ Wih^T, K = 1024
        const ushort_t* a0   = Xg + (size_t)nlo * 1024;
        const ushort_t* a1   = Xg + (size_t)(16 + nlo) * 1024;
        const ushort_t* brow = Wih + (size_t)(n0 + nlo) * 1024;
        for (int k0 = 0; k0 < 1024; k0 += 32) {
            BfTile A0, A1, Bm;
            A0.q[0] = *(const uint4*)(a0 + k0 + aoff);
            A0.q[1] = *(const uint4*)(a0 + k0 + aoff + 16);
            A1.q[0] = *(const uint4*)(a1 + k0 + aoff);
            A1.q[1] = *(const uint4*)(a1 + k0 + aoff + 16);
            Bm.q[0] = *(const uint4*)(brow + k0 + kh * 8);
            Bm.q[1] = *(const uint4*)(brow + k0 + kh * 8 + 16);
            ax0 = __builtin_amdgcn_wmma_f32_16x16x32_bf16(false, A0.v, false, Bm.v, (short)0, ax0, false, false);
            ax1 = __builtin_amdgcn_wmma_f32_16x16x32_bf16(false, A1.v, false, Bm.v, (short)0, ax1, false, false);
        }
    }
    {   // H @ Whh^T, K = 512
        const ushort_t* a0   = Hb + (size_t)nlo * 512;
        const ushort_t* a1   = Hb + (size_t)(16 + nlo) * 512;
        const ushort_t* brow = Whh + (size_t)(n0 + nlo) * 512;
        for (int k0 = 0; k0 < 512; k0 += 32) {
            BfTile A0, A1, Bm;
            A0.q[0] = *(const uint4*)(a0 + k0 + aoff);
            A0.q[1] = *(const uint4*)(a0 + k0 + aoff + 16);
            A1.q[0] = *(const uint4*)(a1 + k0 + aoff);
            A1.q[1] = *(const uint4*)(a1 + k0 + aoff + 16);
            Bm.q[0] = *(const uint4*)(brow + k0 + kh * 8);
            Bm.q[1] = *(const uint4*)(brow + k0 + kh * 8 + 16);
            ah0 = __builtin_amdgcn_wmma_f32_16x16x32_bf16(false, A0.v, false, Bm.v, (short)0, ah0, false, false);
            ah1 = __builtin_amdgcn_wmma_f32_16x16x32_bf16(false, A1.v, false, Bm.v, (short)0, ah1, false, false);
        }
    }

    const int col = n0 + nlo;
    const float bx = b_ih[col], bh = b_hh[col];
    #pragma unroll
    for (int r = 0; r < 8; ++r) {
        const int m = r + kh * 8;                 // C/D layout: row = r + 8*(lane>>4)
        Gx[(size_t)m        * 1536 + col] = ax0[r] + bx;
        Gx[(size_t)(m + 16) * 1536 + col] = ax1[r] + bx;
        Gh[(size_t)m        * 1536 + col] = ah0[r] + bh;
        Gh[(size_t)(m + 16) * 1536 + col] = ah1[r] + bh;
    }
}

// ---------------- step kernel 3: GRU pointwise update ----------------

__global__ __launch_bounds__(256) void gru_update(
    const float* __restrict__ Gx, const float* __restrict__ Gh,
    const float* __restrict__ ctx_f,
    float* __restrict__ h, ushort_t* __restrict__ hbf, ushort_t* __restrict__ ybf)
{
    const int idx = blockIdx.x * 256 + threadIdx.x;   // 0..16383
    const int b = idx >> 9, j = idx & 511;
    const size_t row = (size_t)b * 1536;
    const float xr = Gx[row + j],        hr = Gh[row + j];
    const float xz = Gx[row + 512 + j],  hz = Gh[row + 512 + j];
    const float xn = Gx[row + 1024 + j], hn = Gh[row + 1024 + j];
    const float r = 1.f / (1.f + expf(-(xr + hr)));
    const float z = 1.f / (1.f + expf(-(xz + hz)));
    const float n = tanhf(xn + r * hn);
    const float hp = h[idx];
    const float hnew = (1.f - z) * n + z * hp;
    h[idx]   = hnew;
    hbf[idx] = f2bf(hnew);
    ybf[(size_t)b * 1024 + j]       = f2bf(hnew);
    ybf[(size_t)b * 1024 + 512 + j] = f2bf(ctx_f[idx]);
}

// ---------------- step kernel 4: logits GEMM via WMMA ----------------
// out[:,t,:] = Y[32,1024] @ Wout^T + b_out ; N = 32000 -> 2000 wave tiles

__global__ __launch_bounds__(256) void logits_gemm(
    const ushort_t* __restrict__ Yb,    // [32,1024] bf16
    const ushort_t* __restrict__ Wout,  // [32000,1024] bf16
    const float* __restrict__ bias,     // [32000]
    float* __restrict__ out,            // [B,T,V]
    int t)
{
    const int wave = threadIdx.x >> 5;
    const int lane = threadIdx.x & 31;
    const int n0   = (blockIdx.x * 8 + wave) * 16;    // < 32000
    const int nlo  = lane & 15;
    const int kh   = lane >> 4;
    const int aoff = kh * 8;

    v8f acc0 = {}, acc1 = {};

    const ushort_t* a0   = Yb + (size_t)nlo * 1024;
    const ushort_t* a1   = Yb + (size_t)(16 + nlo) * 1024;
    const ushort_t* brow = Wout + (size_t)(n0 + nlo) * 1024;

    #pragma unroll 2
    for (int k0 = 0; k0 < 1024; k0 += 32) {
        BfTile A0, A1, Bm;
        A0.q[0] = *(const uint4*)(a0 + k0 + aoff);
        A0.q[1] = *(const uint4*)(a0 + k0 + aoff + 16);
        A1.q[0] = *(const uint4*)(a1 + k0 + aoff);
        A1.q[1] = *(const uint4*)(a1 + k0 + aoff + 16);
        Bm.q[0] = *(const uint4*)(brow + k0 + kh * 8);
        Bm.q[1] = *(const uint4*)(brow + k0 + kh * 8 + 16);
        acc0 = __builtin_amdgcn_wmma_f32_16x16x32_bf16(false, A0.v, false, Bm.v, (short)0, acc0, false, false);
        acc1 = __builtin_amdgcn_wmma_f32_16x16x32_bf16(false, A1.v, false, Bm.v, (short)0, acc1, false, false);
    }

    const int col = n0 + nlo;
    const float bv = bias[col];
    #pragma unroll
    for (int r = 0; r < 8; ++r) {
        const int m = r + kh * 8;
        out[((size_t)m        * Tt + t) * Vv + col] = acc0[r] + bv;
        out[((size_t)(m + 16) * Tt + t) * Vv + col] = acc1[r] + bv;
    }
}

// ---------------- host side ----------------

extern "C" void kernel_launch(void* const* d_in, const int* in_sizes, int n_in,
                              void* d_out, int out_size, void* d_ws, size_t ws_size,
                              hipStream_t stream) {
    const float* enc_out  = (const float*)d_in[0];   // [32,128,512]
    const float* enc_hid  = (const float*)d_in[1];   // [1,32,512]
    const int*   targets  = (const int*)  d_in[2];   // [32,128]
    const float* emb_tab  = (const float*)d_in[3];   // [32000,512]
    const float* W_attn   = (const float*)d_in[4];   // [512,512]
    const float* W_ih     = (const float*)d_in[5];   // [1536,1024]
    const float* W_hh     = (const float*)d_in[6];   // [1536,512]
    const float* b_ih     = (const float*)d_in[7];
    const float* b_hh     = (const float*)d_in[8];
    const float* W_out    = (const float*)d_in[9];   // [32000,1024]
    const float* b_out    = (const float*)d_in[10];
    float* out = (float*)d_out;

    // workspace layout (bytes, all regions 256B aligned)
    char* ws = (char*)d_ws;
    size_t off = 0;
    ushort_t* Wout_bf = (ushort_t*)(ws + off); off += (size_t)Vv * 1024 * 2;   // 65,536,000
    ushort_t* Wih_bf  = (ushort_t*)(ws + off); off += (size_t)1536 * 1024 * 2; //  3,145,728
    ushort_t* Whh_bf  = (ushort_t*)(ws + off); off += (size_t)1536 * 512 * 2;  //  1,572,864
    float*    h_buf   = (float*)   (ws + off); off += (size_t)Bb * Hh * 4;
    ushort_t* hbf     = (ushort_t*)(ws + off); off += (size_t)Bb * Hh * 2 + 256;
    ushort_t* xgru_bf = (ushort_t*)(ws + off); off += (size_t)Bb * 2 * Hh * 2;
    ushort_t* ybf     = (ushort_t*)(ws + off); off += (size_t)Bb * 2 * Hh * 2;
    float*    ctx_f   = (float*)   (ws + off); off += (size_t)Bb * Hh * 4;
    float*    Gx      = (float*)   (ws + off); off += (size_t)Bb * 1536 * 4;
    float*    Gh      = (float*)   (ws + off); off += (size_t)Bb * 1536 * 4;

    // one-time preprocessing: bf16 weight copies + hidden init
    cvt_f32_bf16<<<2048, 256, 0, stream>>>(W_out, Wout_bf, Vv * 1024);
    cvt_f32_bf16<<<512,  256, 0, stream>>>(W_ih,  Wih_bf,  1536 * 1024);
    cvt_f32_bf16<<<256,  256, 0, stream>>>(W_hh,  Whh_bf,  1536 * 512);
    init_hidden<<<64, 256, 0, stream>>>(enc_hid, h_buf, hbf);

    for (int t = 0; t < Tt; ++t) {
        attn_step<<<Bb, 256, 0, stream>>>(enc_out, h_buf, targets, emb_tab, W_attn,
                                          xgru_bf, ctx_f, t);
        gates_gemm<<<12, 256, 0, stream>>>(xgru_bf, hbf, Wih_bf, Whh_bf, b_ih, b_hh, Gx, Gh);
        gru_update<<<64, 256, 0, stream>>>(Gx, Gh, ctx_f, h_buf, hbf, ybf);
        logits_gemm<<<250, 256, 0, stream>>>(ybf, Wout_bf, b_out, out, t);
    }

    // hidden state output: out[B*T*V ..]
    copy_hidden_out<<<64, 256, 0, stream>>>(h_buf, out + (size_t)Bb * Tt * Vv);
}
